// PairEmbeddingLayer_19559281066590
// MI455X (gfx1250) — compile-verified, matrix operand order checked by reading, not run on hardware
//
#include <hip/hip_runtime.h>

typedef __attribute__((ext_vector_type(16))) __bf16 v16bf;
typedef __attribute__((ext_vector_type(8)))  float  v8f;

#define E_TOT    100000
#define NODE_D   128
#define CTX_D    256
#define OUT_D    256
#define IN_D     384
#define NCOEF    7
#define M_TILE   64
#define MF       4          // M fragments of 16
#define THREADS  256
#define KT_CTX   8          // 256/32
#define KT_BASE  12         // 384/32
#define KT_SPL   84         // 2688/32
#define NT_CTX   8          // 128/16
#define NT_OUT   16         // 256/16

// workspace layout in bf16 (ushort) elements
#define WS_CTX_OFF   0
#define WS_CTX_ELEMS (KT_CTX*NT_CTX*512)     // 32768
#define WS_BASE_OFF  (WS_CTX_OFF + WS_CTX_ELEMS)
#define WS_BASE_ELEMS (KT_BASE*NT_OUT*512)   // 98304
#define WS_SPL_OFF   (WS_BASE_OFF + WS_BASE_ELEMS)
#define WS_SPL_ELEMS (KT_SPL*NT_OUT*512)     // 688128

#define XS_BYTES   (M_TILE*IN_D*4)           // 98304
#define FB_KT      12
#define SMEM_BYTES (XS_BYTES + FB_KT*MF*512*2) // 147456

__device__ __forceinline__ unsigned short f2bf(float f) {
    union { float f; unsigned u; } x; x.f = f;
    unsigned u = x.u + (0x7FFFu + ((x.u >> 16) & 1u));  // RNE
    return (unsigned short)(u >> 16);
}

// position of element (k,n) inside packed B-fragment stream (NT n-tiles wide)
__device__ __forceinline__ int bpos(int k, int n, int NT) {
    int kt = k >> 5, kk = k & 31;
    int t = kk >> 4, r = kk & 15, g = r >> 3, rr = r & 7;
    int v = t*4 + (rr >> 1), h = rr & 1;
    int lane = (n & 15) + (g << 4);
    return (((kt*NT + (n >> 4))*32 + lane) << 4) + v*2 + h;
}

// position of element (k,m) inside LDS A-fragment buffer (MF m-frags wide)
__device__ __forceinline__ int apos(int k, int m) {
    int kt = k >> 5, kk = k & 31;
    int t = kk >> 4, r = kk & 15, g = r >> 3, rr = r & 7;
    int v = t*4 + (rr >> 1), h = rr & 1;
    int lane = (m & 15) + (g << 4);
    return (((kt*MF + (m >> 4))*32 + lane) << 4) + v*2 + h;
}

__global__ __launch_bounds__(THREADS) void pack_ctx_w(const float* __restrict__ w,
                                                      unsigned short* __restrict__ dst) {
    int idx = blockIdx.x*THREADS + threadIdx.x;
    if (idx >= CTX_D*NODE_D) return;
    int k = idx / NODE_D, n = idx % NODE_D;           // B[k][n] = ctx_w[n][k]
    dst[bpos(k, n, NT_CTX)] = f2bf(w[n*CTX_D + k]);
}

__global__ __launch_bounds__(THREADS) void pack_base_w(const float* __restrict__ w,
                                                       unsigned short* __restrict__ dst) {
    int idx = blockIdx.x*THREADS + threadIdx.x;
    if (idx >= IN_D*OUT_D) return;
    int k = idx / OUT_D, n = idx % OUT_D;             // B[k][n] = base_w[n][k]
    dst[bpos(k, n, NT_OUT)] = f2bf(w[n*IN_D + k]);
}

__global__ __launch_bounds__(THREADS) void pack_spl_w(const float* __restrict__ w,
                                                      const float* __restrict__ sc,
                                                      unsigned short* __restrict__ dst) {
    int idx = blockIdx.x*THREADS + threadIdx.x;
    if (idx >= KT_SPL*32*OUT_D) return;
    int k = idx / OUT_D, n = idx % OUT_D;             // k = i*7 + c
    int i = k / NCOEF, c = k - i*NCOEF;
    float v = w[(n*IN_D + i)*NCOEF + c] * sc[n*IN_D + i];
    dst[bpos(k, n, NT_OUT)] = f2bf(v);
}

__global__ __launch_bounds__(THREADS) void pair_kan_kernel(
    const float* __restrict__ node_emb, const float* __restrict__ ctx_emb,
    const int*   __restrict__ pairs,    const float* __restrict__ ctx_b,
    float* __restrict__ out,
    const unsigned short* __restrict__ Bctx,
    const unsigned short* __restrict__ Bbase,
    const unsigned short* __restrict__ Bspl)
{
    extern __shared__ char smem[];
    float* xs = (float*)smem;                               // [64][384] fp32
    unsigned short* fb = (unsigned short*)(smem + XS_BYTES);// A-fragment buffer
    __shared__ int prIdx[M_TILE][2];

    const int tid  = threadIdx.x;
    const int lane = tid & 31;
    const int wave = tid >> 5;
    const int ebase = blockIdx.x * M_TILE;

    if (tid < M_TILE*2) {
        int m = tid >> 1, w = tid & 1;
        int e = ebase + m; if (e >= E_TOT) e = E_TOT - 1;
        prIdx[m][w] = pairs[e*2 + w];
    }
    __syncthreads();

    // gather subj/obj rows into xs[:,0:128] and xs[:,128:256]
    for (int idx = tid; idx < M_TILE*NODE_D; idx += THREADS) {
        int m = idx >> 7, c = idx & 127;
        xs[m*IN_D + c]          = node_emb[prIdx[m][0]*NODE_D + c];
        xs[m*IN_D + NODE_D + c] = node_emb[prIdx[m][1]*NODE_D + c];
    }
    // stage ctx-embedding A-fragments (bf16), K = 256
    for (int idx = tid; idx < M_TILE*CTX_D; idx += THREADS) {
        int m = idx >> 8, k = idx & 255;
        int e = ebase + m; if (e >= E_TOT) e = E_TOT - 1;
        fb[apos(k, m)] = f2bf(ctx_emb[e*CTX_D + k]);
    }
    __syncthreads();

    // ---- ctx GEMM: (64x256)x(256x128); wave owns one 16-wide n-tile ----
    {
        v8f cacc[MF] = {};
        const int nt = wave;
        #pragma unroll
        for (int kt = 0; kt < KT_CTX; ++kt) {
            v16bf b = *(const v16bf*)(Bctx + (((kt*NT_CTX + nt)*32 + lane) << 4));
            #pragma unroll
            for (int mf = 0; mf < MF; ++mf) {
                v16bf a = *(const v16bf*)(fb + (((kt*MF + mf)*32 + lane) << 4));
                cacc[mf] = __builtin_amdgcn_wmma_f32_16x16x32_bf16(
                    false, a, false, b, (short)0, cacc[mf], false, false);
            }
        }
        int n = nt*16 + (lane & 15);
        float bias = ctx_b[n];
        int g = lane >> 4;
        #pragma unroll
        for (int mf = 0; mf < MF; ++mf)
            #pragma unroll
            for (int v = 0; v < 8; ++v)
                xs[(mf*16 + v + 8*g)*IN_D + 2*NODE_D + n] = cacc[mf][v] + bias;
    }
    __syncthreads();

    v8f acc[MF][2] = {};
    const int nt0 = wave*2;

    // ---- base GEMM: silu(x) (64x384) x (384x256) ----
    for (int idx = tid; idx < M_TILE*IN_D; idx += THREADS) {
        int m = idx / IN_D, k = idx - m*IN_D;
        float v = xs[m*IN_D + k];
        fb[apos(k, m)] = f2bf(v / (1.0f + __expf(-v)));
    }
    __syncthreads();
    for (int kt = 0; kt < KT_BASE; ++kt) {
        v16bf a[MF];
        #pragma unroll
        for (int mf = 0; mf < MF; ++mf)
            a[mf] = *(const v16bf*)(fb + (((kt*MF + mf)*32 + lane) << 4));
        #pragma unroll
        for (int j = 0; j < 2; ++j) {
            v16bf b = *(const v16bf*)(Bbase + (((kt*NT_OUT + nt0 + j)*32 + lane) << 4));
            #pragma unroll
            for (int mf = 0; mf < MF; ++mf)
                acc[mf][j] = __builtin_amdgcn_wmma_f32_16x16x32_bf16(
                    false, a[mf], false, b, (short)0, acc[mf][j], false, false);
        }
    }
    __syncthreads();

    // ---- spline GEMM: bsp (64x2688) x (2688x256), 12 chunks of 32 dims ----
    for (int ch = 0; ch < 12; ++ch) {
        for (int idx = tid; idx < M_TILE*32; idx += THREADS) {
            int m = idx >> 5, ii = idx & 31;
            float x = xs[m*IN_D + ch*32 + ii];
            // uniform cubic B-spline, knots -2.5..2.5 step 0.5
            float s  = (x + 2.5f) * 2.0f;
            float fj = floorf(s);
            int   j0 = (int)fj;
            float f  = s - fj;
            float f2 = f*f, f3 = f2*f, q = 1.0f - f;
            float p0 = f3 * (1.0f/6.0f);
            float p1 = (-3.0f*f3 + 3.0f*f2 + 3.0f*f + 1.0f) * (1.0f/6.0f);
            float p2 = ( 3.0f*f3 - 6.0f*f2 + 4.0f) * (1.0f/6.0f);
            float p3 = q*q*q * (1.0f/6.0f);
            bool inr = (j0 >= 0) && (j0 <= 9);
            #pragma unroll
            for (int c = 0; c < NCOEF; ++c) {
                int r = j0 - c;
                float val = 0.0f;
                if (inr) {
                    if      (r == 0) val = p0;
                    else if (r == 1) val = p1;
                    else if (r == 2) val = p2;
                    else if (r == 3) val = p3;
                }
                fb[apos(ii*NCOEF + c, m)] = f2bf(val);
            }
        }
        __syncthreads();
        #pragma unroll
        for (int kt = 0; kt < 7; ++kt) {
            v16bf a[MF];
            #pragma unroll
            for (int mf = 0; mf < MF; ++mf)
                a[mf] = *(const v16bf*)(fb + (((kt*MF + mf)*32 + lane) << 4));
            #pragma unroll
            for (int j = 0; j < 2; ++j) {
                v16bf b = *(const v16bf*)(Bspl + ((((ch*7 + kt)*NT_OUT + nt0 + j)*32 + lane) << 4));
                #pragma unroll
                for (int mf = 0; mf < MF; ++mf)
                    acc[mf][j] = __builtin_amdgcn_wmma_f32_16x16x32_bf16(
                        false, a[mf], false, b, (short)0, acc[mf][j], false, false);
            }
        }
        __syncthreads();
    }

    // ---- store C (fp32) ----
    #pragma unroll
    for (int j = 0; j < 2; ++j) {
        int n = (nt0 + j)*16 + (lane & 15);
        #pragma unroll
        for (int mf = 0; mf < MF; ++mf)
            #pragma unroll
            for (int v = 0; v < 8; ++v) {
                int m = mf*16 + v + 8*(lane >> 4);
                int e = ebase + m;
                if (e < E_TOT) out[e*OUT_D + n] = acc[mf][j][v];
            }
    }
}

extern "C" void kernel_launch(void* const* d_in, const int* in_sizes, int n_in,
                              void* d_out, int out_size, void* d_ws, size_t ws_size,
                              hipStream_t stream) {
    const float* node_emb = (const float*)d_in[0];
    const float* ctx_emb  = (const float*)d_in[1];
    const int*   pairs    = (const int*)d_in[2];
    const float* ctx_w    = (const float*)d_in[3];
    const float* ctx_b    = (const float*)d_in[4];
    const float* base_w   = (const float*)d_in[5];
    const float* spl_w    = (const float*)d_in[6];
    const float* spl_s    = (const float*)d_in[7];
    float* out = (float*)d_out;

    unsigned short* ws    = (unsigned short*)d_ws;
    unsigned short* Bctx  = ws + WS_CTX_OFF;
    unsigned short* Bbase = ws + WS_BASE_OFF;
    unsigned short* Bspl  = ws + WS_SPL_OFF;

    pack_ctx_w <<<(WS_CTX_ELEMS  + THREADS-1)/THREADS, THREADS, 0, stream>>>(ctx_w, Bctx);
    pack_base_w<<<(WS_BASE_ELEMS + THREADS-1)/THREADS, THREADS, 0, stream>>>(base_w, Bbase);
    pack_spl_w <<<(WS_SPL_ELEMS  + THREADS-1)/THREADS, THREADS, 0, stream>>>(spl_w, spl_s, Bspl);

    (void)hipFuncSetAttribute(reinterpret_cast<const void*>(&pair_kan_kernel),
                              hipFuncAttributeMaxDynamicSharedMemorySize, SMEM_BYTES);

    int nblocks = (E_TOT + M_TILE - 1) / M_TILE;   // 1563
    pair_kan_kernel<<<nblocks, THREADS, SMEM_BYTES, stream>>>(
        node_emb, ctx_emb, pairs, ctx_b, out, Bctx, Bbase, Bspl);
}
